// ImageDomainMRIUFormer_68848325755326
// MI455X (gfx1250) — compile-verified, matrix operand order checked by reading, not run on hardware
//
#include <hip/hip_runtime.h>

typedef __bf16 bf16;
typedef __attribute__((ext_vector_type(16))) __bf16 v16bf;
typedef __attribute__((ext_vector_type(8)))  __bf16 v8bf;
typedef __attribute__((ext_vector_type(8)))  float  v8f;
typedef __attribute__((ext_vector_type(4)))  int    v4i;
typedef __attribute__((ext_vector_type(4)))  unsigned u32x4;
typedef __attribute__((ext_vector_type(8)))  int    i32x8;
typedef __attribute__((ext_vector_type(4)))  int    i32x4;

#define DEV __device__ __forceinline__

// ---------------------------------------------------------------- async global->LDS (guarded)
#if __has_builtin(__builtin_amdgcn_global_load_async_to_lds_b128)
#define HAVE_ASYNC_LDS 1
DEV void async_cp16(const void* g, void* l) {
  __builtin_amdgcn_global_load_async_to_lds_b128((v4i*)g, (v4i*)l, 0, 0);
}
DEV void async_wait0() {
#if __has_builtin(__builtin_amdgcn_s_wait_asynccnt)
  __builtin_amdgcn_s_wait_asynccnt(0);
#else
  asm volatile("s_wait_asynccnt 0x0" ::: "memory");
#endif
}
#else
#define HAVE_ASYNC_LDS 0
#endif

// ---------------------------------------------------------------- TDM (guarded)
#if HAVE_ASYNC_LDS && __has_builtin(__builtin_amdgcn_tensor_load_to_lds)
#define HAVE_TDM 1
DEV unsigned lds_off(void* p) {
  return (unsigned)(unsigned long long)(__attribute__((address_space(3))) char*)p;
}
// 2D TDM: load tile (32 rows x 32 bf16) from row-major (rows x K) matrix at gsrc into LDS.
DEV void tdm_load_b_tile(const bf16* gsrc, unsigned ldsOff, int K) {
  unsigned long long ga = (unsigned long long)(size_t)gsrc;
  u32x4 g0;
  g0[0] = 1u;                                   // count=1, user mode
  g0[1] = ldsOff;                               // lds_addr (bytes)
  g0[2] = (unsigned)ga;                         // global_addr[31:0]
  g0[3] = (unsigned)(ga >> 32) | (2u << 30);    // global_addr[56:32] | type=2
  i32x8 g1;
  g1[0] = (int)(1u << 16);                      // workgroup_mask=0, data_size=1 (2B)
  g1[1] = (int)((unsigned)K << 16);             // tensor_dim0[15:0] = K (K < 65536)
  g1[2] = (int)(32u << 16);                     // tensor_dim0 hi=0, tensor_dim1[15:0]=32
  g1[3] = (int)(32u << 16);                     // tensor_dim1 hi=0, tile_dim0=32
  g1[4] = 32;                                   // tile_dim1=32, tile_dim2=0
  g1[5] = K;                                    // tensor_dim0_stride lo = K
  g1[6] = 0;                                    // stride hi, dim1_stride lo
  g1[7] = 0;
  i32x4 z4 = {0, 0, 0, 0};
  i32x8 z8 = {0, 0, 0, 0, 0, 0, 0, 0};
  __builtin_amdgcn_tensor_load_to_lds(g0, g1, z4, z4, z8, 0);
}
DEV void tensor_wait0() {
#if __has_builtin(__builtin_amdgcn_s_wait_tensorcnt)
  __builtin_amdgcn_s_wait_tensorcnt(0);
#else
  asm volatile("s_wait_tensorcnt 0x0" ::: "memory");
#endif
}
DEV void tensor_wait1() {
#if __has_builtin(__builtin_amdgcn_s_wait_tensorcnt)
  __builtin_amdgcn_s_wait_tensorcnt(1);
#else
  asm volatile("s_wait_tensorcnt 0x1" ::: "memory");
#endif
}
#else
#define HAVE_TDM 0
#endif

#if HAVE_ASYNC_LDS
DEV void async_wait_prev() {   // allow next tile's loads to remain in flight
#if HAVE_TDM
#if __has_builtin(__builtin_amdgcn_s_wait_asynccnt)
  __builtin_amdgcn_s_wait_asynccnt(4);          // 4 A-loads outstanding for next tile
#else
  asm volatile("s_wait_asynccnt 0x4" ::: "memory");
#endif
#else
#if __has_builtin(__builtin_amdgcn_s_wait_asynccnt)
  __builtin_amdgcn_s_wait_asynccnt(5);          // 4 A + 1 B outstanding for next tile
#else
  asm volatile("s_wait_asynccnt 0x5" ::: "memory");
#endif
#endif
}
#endif

// ---------------------------------------------------------------- WMMA helpers
DEV v8f wmma_bf16(v16bf a, v16bf b, v8f c) {
  return __builtin_amdgcn_wmma_f32_16x16x32_bf16(false, a, false, b, (short)0, c, false, false);
}

// Fragment (16 rows x 32 K bf16) from a row-major tile, ld elements (multiple of 8).
// Serves A (rows = M) and B-as-transposed (rows = N). Contiguous 16B LDS loads.
DEV v16bf frag_a(const bf16* p, int ld, int lane) {
  int r = lane & 15, kb = (lane >> 4) << 3;
  const bf16* row = p + r * ld + kb;
  v8bf lo = *(const v8bf*)(row);
  v8bf hi = *(const v8bf*)(row + 16);
  v16bf f;
#pragma unroll
  for (int i = 0; i < 8; ++i) { f[i] = lo[i]; f[i + 8] = hi[i]; }
  return f;
}

DEV float gelu_f(float x) { return 0.5f * x * (1.0f + erff(x * 0.70710678118654752f)); }

// ---------------------------------------------------------------- generic GEMM
// out(MxN,f32) = A(MxK,bf16) @ B(KxN) with B supplied TRANSPOSED as Bt(NxK,bf16).
// grid (M/128, N/32), 128 threads (4 waves); wave w rows [32w,32w+32): 2x2 WMMA tiles.
// Double-buffered async global->LDS pipeline; B tile via TDM when available.
__global__ __launch_bounds__(128) void k_gemm(const bf16* __restrict__ A, const bf16* __restrict__ Bt,
                                              const float* __restrict__ bias, const float* __restrict__ resid,
                                              float* __restrict__ out, int M, int N, int K, int act) {
  __shared__ __align__(16) bf16 As[2][128 * 32];
  __shared__ __align__(16) bf16 Bs[2][32 * 32];   // N x K layout
  int tid = threadIdx.x, lane = tid & 31, wave = tid >> 5;
  int m0 = blockIdx.x * 128, n0 = blockIdx.y * 32;
  v8f acc[2][2];
#pragma unroll
  for (int a = 0; a < 2; ++a)
#pragma unroll
    for (int b = 0; b < 2; ++b) acc[a][b] = (v8f){0, 0, 0, 0, 0, 0, 0, 0};

#if HAVE_ASYNC_LDS
  auto issueA = [&](int k0, int buf) {
#pragma unroll
    for (int i = 0; i < 4; ++i) {                 // A tile: 128x32 = 512 x 16B
      int idx = tid + i * 128, row = idx >> 2, c4 = idx & 3;
      async_cp16(A + (size_t)(m0 + row) * K + k0 + c4 * 8, &As[buf][row * 32 + c4 * 8]);
    }
  };
  auto issueB = [&](int k0, int buf) {
#if HAVE_TDM
    if (wave == 0) tdm_load_b_tile(Bt + (size_t)n0 * K + k0, lds_off(&Bs[buf][0]), K);
#else
    int n = tid >> 2, c4 = tid & 3;
    async_cp16(Bt + (size_t)(n0 + n) * K + k0 + c4 * 8, &Bs[buf][n * 32 + c4 * 8]);
#endif
  };
  int nk = K >> 5;
  issueA(0, 0); issueB(0, 0);
  for (int kt = 0; kt < nk; ++kt) {
    int cur = kt & 1;
    bool more = (kt + 1 < nk);
    if (more) { issueA((kt + 1) * 32, cur ^ 1); issueB((kt + 1) * 32, cur ^ 1); }
    if (more) async_wait_prev(); else async_wait0();
#if HAVE_TDM
    if (wave == 0) { if (more) tensor_wait1(); else tensor_wait0(); }
#endif
    __syncthreads();
    const bf16* Ab = As[cur];
    const bf16* Bb = Bs[cur];
    v16bf a0 = frag_a(Ab + (wave * 32) * 32, 32, lane);
    v16bf a1 = frag_a(Ab + (wave * 32 + 16) * 32, 32, lane);
    v16bf b0 = frag_a(Bb, 32, lane);
    v16bf b1 = frag_a(Bb + 16 * 32, 32, lane);
    acc[0][0] = wmma_bf16(a0, b0, acc[0][0]);
    acc[0][1] = wmma_bf16(a0, b1, acc[0][1]);
    acc[1][0] = wmma_bf16(a1, b0, acc[1][0]);
    acc[1][1] = wmma_bf16(a1, b1, acc[1][1]);
    __syncthreads();
  }
#else
  for (int k0 = 0; k0 < K; k0 += 32) {
#pragma unroll
    for (int i = 0; i < 4; ++i) {
      int idx = tid + i * 128, row = idx >> 2, c4 = idx & 3;
      ((uint4*)As[0])[idx] = *(const uint4*)(A + (size_t)(m0 + row) * K + k0 + c4 * 8);
    }
    {
      int n = tid >> 2, c4 = tid & 3;
      ((uint4*)Bs[0])[tid] = *(const uint4*)(Bt + (size_t)(n0 + n) * K + k0 + c4 * 8);
    }
    if (k0 + 32 < K) {
      int row = tid >> 2, c4 = tid & 3;
      __builtin_prefetch(A + (size_t)(m0 + row) * K + k0 + 32 + c4 * 8, 0, 1);
      __builtin_prefetch(Bt + (size_t)(n0 + row) * K + k0 + 32 + c4 * 8, 0, 1);
    }
    __syncthreads();
    v16bf a0 = frag_a(As[0] + (wave * 32) * 32, 32, lane);
    v16bf a1 = frag_a(As[0] + (wave * 32 + 16) * 32, 32, lane);
    v16bf b0 = frag_a(Bs[0], 32, lane);
    v16bf b1 = frag_a(Bs[0] + 16 * 32, 32, lane);
    acc[0][0] = wmma_bf16(a0, b0, acc[0][0]);
    acc[0][1] = wmma_bf16(a0, b1, acc[0][1]);
    acc[1][0] = wmma_bf16(a1, b0, acc[1][0]);
    acc[1][1] = wmma_bf16(a1, b1, acc[1][1]);
    __syncthreads();
  }
#endif
  int msub = (lane >> 4) << 3;
  int nc = n0 + (lane & 15);
#pragma unroll
  for (int mt = 0; mt < 2; ++mt) {
#pragma unroll
    for (int r = 0; r < 8; ++r) {
      int mm = m0 + wave * 32 + mt * 16 + msub + r;
      float v0 = acc[mt][0][r], v1 = acc[mt][1][r];
      if (bias) { v0 += bias[nc]; v1 += bias[nc + 16]; }
      if (act == 1) { v0 = gelu_f(v0); v1 = gelu_f(v1); }
      if (resid) { v0 += resid[(size_t)mm * N + nc]; v1 += resid[(size_t)mm * N + nc + 16]; }
      out[(size_t)mm * N + nc] = v0;
      out[(size_t)mm * N + nc + 16] = v1;
    }
  }
}

// ---------------------------------------------------------------- window attention
DEV int region3(int v, int res) { return v < res - 8 ? 0 : (v < res - 4 ? 1 : 2); }

// one block = one (window, head). Q,K,V 64x32, head_dim 32. grid (nW, heads), 128 thr.
// Output written directly as bf16 (feeds the projection GEMM).
__global__ __launch_bounds__(128) void k_attn(const float* __restrict__ qkv, const float* __restrict__ rpb,
                                              bf16* __restrict__ out, int C, int heads, int res, int shift) {
  __shared__ __align__(16) bf16 Qs[64 * 32];
  __shared__ __align__(16) bf16 Ks[64 * 32];
  __shared__ __align__(16) bf16 Vt[32 * 72];   // V transposed (d rows, p cols), padded ld=72
  __shared__ __align__(16) bf16 Ps[64 * 64];
  int win = blockIdx.x, head = blockIdx.y;
  int nw = res >> 3, wy = win / nw, wx = win % nw;
  int tid = threadIdx.x, lane = tid & 31, wave = tid >> 5;
  int C3 = 3 * C;
  const float scl = 0.17677669529663689f;  // 32^-0.5
  for (int i = tid; i < 2048; i += 128) {
    int p = i >> 5, d = i & 31;
    const float* base = qkv + (size_t)(win * 64 + p) * C3 + head * 32 + d;
    Qs[i] = (bf16)(base[0] * scl);
    Ks[i] = (bf16)(base[C]);
    Vt[d * 72 + p] = (bf16)(base[2 * C]);
  }
  __syncthreads();
  int qbase = wave * 16;
  v16bf qa = frag_a(Qs + qbase * 32, 32, lane);
  v8f s[4];
#pragma unroll
  for (int nt = 0; nt < 4; ++nt) {           // S = Q @ K^T (K^T loads with A-pattern)
    v16bf kb = frag_a(Ks + nt * 16 * 32, 32, lane);
    v8f z = {0, 0, 0, 0, 0, 0, 0, 0};
    s[nt] = wmma_bf16(qa, kb, z);
  }
  int mloc = (lane >> 4) << 3;
#pragma unroll
  for (int nt = 0; nt < 4; ++nt) {           // + relative position bias + swin mask
    int kcol = nt * 16 + (lane & 15);
    int ki = kcol >> 3, kj = kcol & 7;
#pragma unroll
    for (int r = 0; r < 8; ++r) {
      int q = qbase + mloc + r;
      int qi = q >> 3, qj = q & 7;
      float v = rpb[((qi - ki + 7) * 15 + (qj - kj + 7)) * heads + head];
      if (shift) {
        int la = region3(wy * 8 + qi, res) * 3 + region3(wx * 8 + qj, res);
        int lb = region3(wy * 8 + ki, res) * 3 + region3(wx * 8 + kj, res);
        if (la != lb) v -= 100.0f;
      }
      s[nt][r] += v;
    }
  }
  float rmax[8], rinv[8];
#pragma unroll
  for (int r = 0; r < 8; ++r) {              // row max over 64 keys (4 tiles x 16 lanes/half)
    float m = fmaxf(fmaxf(s[0][r], s[1][r]), fmaxf(s[2][r], s[3][r]));
#pragma unroll
    for (int o = 1; o < 16; o <<= 1) m = fmaxf(m, __shfl_xor(m, o, 32));
    rmax[r] = m;
  }
#pragma unroll
  for (int nt = 0; nt < 4; ++nt)
#pragma unroll
    for (int r = 0; r < 8; ++r) s[nt][r] = __expf(s[nt][r] - rmax[r]);
#pragma unroll
  for (int r = 0; r < 8; ++r) {
    float t = s[0][r] + s[1][r] + s[2][r] + s[3][r];
#pragma unroll
    for (int o = 1; o < 16; o <<= 1) t += __shfl_xor(t, o, 32);
    rinv[r] = 1.0f / t;
  }
#pragma unroll
  for (int nt = 0; nt < 4; ++nt)
#pragma unroll
    for (int r = 0; r < 8; ++r)
      Ps[(qbase + mloc + r) * 64 + nt * 16 + (lane & 15)] = (bf16)(s[nt][r] * rinv[r]);
  __syncthreads();
  v8f o0 = {0, 0, 0, 0, 0, 0, 0, 0}, o1 = {0, 0, 0, 0, 0, 0, 0, 0};
#pragma unroll
  for (int kt = 0; kt < 2; ++kt) {           // O = P @ V, K=64 in two 32-steps, V^T in LDS
    v16bf pa  = frag_a(Ps + qbase * 64 + kt * 32, 64, lane);
    v16bf vb0 = frag_a(Vt + kt * 32, 72, lane);
    v16bf vb1 = frag_a(Vt + 16 * 72 + kt * 32, 72, lane);
    o0 = wmma_bf16(pa, vb0, o0);
    o1 = wmma_bf16(pa, vb1, o1);
  }
  int n = lane & 15;
#pragma unroll
  for (int r = 0; r < 8; ++r) {
    int q = qbase + mloc + r;
    size_t ro = (size_t)(win * 64 + q) * C + head * 32;
    out[ro + n] = (bf16)o0[r];
    out[ro + 16 + n] = (bf16)o1[r];
  }
}

// ---------------------------------------------------------------- FFT (256-pt, inverse, LDS)
DEV void fft256_inv(float* re, float* im) {
#pragma unroll
  for (int s = 1; s <= 8; ++s) {
    __syncthreads();
    int mh = 1 << (s - 1);
    int t = threadIdx.x;
    int grp = t >> (s - 1), pos = t & (mh - 1);
    int i0 = (grp << s) + pos, i1 = i0 + mh;
    float ang = 6.283185307179586f * (float)pos / (float)(1 << s);
    float wr = __cosf(ang), wi = __sinf(ang);
    float vr = re[i1] * wr - im[i1] * wi;
    float vi = re[i1] * wi + im[i1] * wr;
    float ur = re[i0], ui = im[i0];
    re[i0] = ur + vr; im[i0] = ui + vi;
    re[i1] = ur - vr; im[i1] = ui - vi;
  }
  __syncthreads();
}

__global__ __launch_bounds__(128) void k_ifft_x(const float* __restrict__ in, float* __restrict__ out) {
  __shared__ float re[256], im[256];
  const float* src = in + (size_t)blockIdx.x * 512;
  float* dst = out + (size_t)blockIdx.x * 512;
  for (int i = threadIdx.x; i < 256; i += 128) {
    int j = (i + 128) & 255;                 // ifftshift
    int rv = __brev((unsigned)i) >> 24;      // bit-reversed load (DIT)
    re[rv] = src[j * 2]; im[rv] = src[j * 2 + 1];
  }
  fft256_inv(re, im);
  for (int i = threadIdx.x; i < 256; i += 128) {
    int j = (i + 128) & 255;                 // fftshift
    dst[i * 2] = re[j] * 0.0625f; dst[i * 2 + 1] = im[j] * 0.0625f;  // ortho 1/sqrt(256)
  }
}

__global__ __launch_bounds__(128) void k_ifft_y(const float* __restrict__ in, float* __restrict__ out) {
  __shared__ float re[256], im[256];
  int coil = blockIdx.x >> 8, x = blockIdx.x & 255;
  const float* src = in + ((size_t)coil * 65536 + x) * 2;
  float* dst = out + ((size_t)coil * 65536 + x) * 2;
  for (int i = threadIdx.x; i < 256; i += 128) {
    int j = (i + 128) & 255;
    int rv = __brev((unsigned)i) >> 24;
    re[rv] = src[(size_t)j * 512]; im[rv] = src[(size_t)j * 512 + 1];
  }
  fft256_inv(re, im);
  for (int i = threadIdx.x; i < 256; i += 128) {
    int j = (i + 128) & 255;
    dst[(size_t)i * 512] = re[j] * 0.0625f; dst[(size_t)i * 512 + 1] = im[j] * 0.0625f;
  }
}

// ---------------------------------------------------------------- elementwise / glue
__global__ void k_coil_reduce(const float* __restrict__ ci, const float* __restrict__ sm,
                              float* __restrict__ img) {
  int i = blockIdx.x * 256 + threadIdx.x;
  if (i >= 65536) return;
  float re = 0.f, im = 0.f;
  for (int co = 0; co < 12; ++co) {
    const float* c = ci + ((size_t)co * 65536 + i) * 2;
    const float* s = sm + ((size_t)co * 65536 + i) * 2;
    re += s[0] * c[0] + s[1] * c[1];
    im += s[0] * c[1] - s[1] * c[0];
  }
  img[i * 2] = re; img[i * 2 + 1] = im;
}

__global__ __launch_bounds__(1024) void k_stats(const float* __restrict__ img, float* __restrict__ st) {
  __shared__ float sh[1024];
  float s0 = 0, s1 = 0, q0 = 0, q1 = 0;
  for (int i = threadIdx.x; i < 65536; i += 1024) {
    float a = img[i * 2], b = img[i * 2 + 1];
    s0 += a; q0 += a * a; s1 += b; q1 += b * b;
  }
  float vals[4] = {s0, s1, q0, q1}, red[4];
  for (int v = 0; v < 4; ++v) {
    __syncthreads();
    sh[threadIdx.x] = vals[v];
    __syncthreads();
    for (int s = 512; s > 0; s >>= 1) {
      if (threadIdx.x < s) sh[threadIdx.x] += sh[threadIdx.x + s];
      __syncthreads();
    }
    red[v] = sh[0];
  }
  if (threadIdx.x == 0) {
    float n = 65536.f;
    float m0 = red[0] / n, m1 = red[1] / n;
    st[0] = m0; st[1] = m1;
    st[2] = sqrtf((red[2] - n * m0 * m0) / (n - 1.f));   // ddof = 1
    st[3] = sqrtf((red[3] - n * m1 * m1) / (n - 1.f));
  }
}

__global__ void k_xn(const float* __restrict__ img, const float* __restrict__ st, float* __restrict__ xn) {
  int i = blockIdx.x * 256 + threadIdx.x;
  if (i >= 131072) return;
  int c = i & 1;
  xn[i] = (img[i] - st[c]) / st[2 + c];
}

__global__ void k_conv_in(const float* __restrict__ xn, const float* __restrict__ w,
                          const float* __restrict__ b, float* __restrict__ out) {
  int i = blockIdx.x * 256 + threadIdx.x;   // 65536 * 32
  if (i >= 65536 * 32) return;
  int c = i & 31, pix = i >> 5;
  int y = pix >> 8, x = pix & 255;
  float acc = b[c];
  for (int ky = 0; ky < 3; ++ky) {
    int iy = y + ky - 1; if (iy < 0 || iy > 255) continue;
    for (int kx = 0; kx < 3; ++kx) {
      int ix = x + kx - 1; if (ix < 0 || ix > 255) continue;
      const float* px = xn + ((size_t)iy * 256 + ix) * 2;
      const float* wp = w + ((ky * 3 + kx) * 2) * 32 + c;
      acc += px[0] * wp[0] + px[1] * wp[32];
    }
  }
  out[i] = acc > 0.f ? acc : 0.01f * acc;
}

__global__ __launch_bounds__(256) void k_layernorm(const float* __restrict__ x, const float* __restrict__ g,
                                                   const float* __restrict__ b, float* __restrict__ out,
                                                   int L, int C) {
  int wave = threadIdx.x >> 5, lane = threadIdx.x & 31;
  int row = blockIdx.x * 8 + wave;
  if (row >= L) return;
  const float* xr = x + (size_t)row * C;
  float s = 0.f, q = 0.f;
  for (int c = lane; c < C; c += 32) { float v = xr[c]; s += v; q += v * v; }
  for (int o = 1; o < 32; o <<= 1) { s += __shfl_xor(s, o, 32); q += __shfl_xor(q, o, 32); }
  float m = s / C;
  float inv = rsqrtf(q / C - m * m + 1e-5f);
  float* orow = out + (size_t)row * C;
  for (int c = lane; c < C; c += 32) orow[c] = (xr[c] - m) * inv * g[c] + b[c];
}

// layernorm emitting bf16 (feeds a GEMM A operand directly)
__global__ __launch_bounds__(256) void k_layernorm_b(const float* __restrict__ x, const float* __restrict__ g,
                                                     const float* __restrict__ b, bf16* __restrict__ out,
                                                     int L, int C) {
  int wave = threadIdx.x >> 5, lane = threadIdx.x & 31;
  int row = blockIdx.x * 8 + wave;
  if (row >= L) return;
  const float* xr = x + (size_t)row * C;
  float s = 0.f, q = 0.f;
  for (int c = lane; c < C; c += 32) { float v = xr[c]; s += v; q += v * v; }
  for (int o = 1; o < 32; o <<= 1) { s += __shfl_xor(s, o, 32); q += __shfl_xor(q, o, 32); }
  float m = s / C;
  float inv = rsqrtf(q / C - m * m + 1e-5f);
  bf16* orow = out + (size_t)row * C;
  for (int c = lane; c < C; c += 32) orow[c] = (bf16)((xr[c] - m) * inv * g[c] + b[c]);
}

__global__ void k_f2b(const float* __restrict__ a, bf16* __restrict__ o, int n) {
  int i = blockIdx.x * 256 + threadIdx.x;
  if (i < n) o[i] = (bf16)a[i];
}

// convert + transpose: w (K x N) f32 -> o (N x K) bf16
__global__ void k_f2bt(const float* __restrict__ w, bf16* __restrict__ o, int K, int N, int total) {
  int i = blockIdx.x * 256 + threadIdx.x;
  if (i >= total) return;
  int k = i % K, n = i / K;
  o[i] = (bf16)w[(size_t)k * N + n];
}

__global__ void k_winpart(const float* __restrict__ x, bf16* __restrict__ out, int res, int C, int shift, int total) {
  int i = blockIdx.x * 256 + threadIdx.x;
  if (i >= total) return;
  int c = i % C, t = i / C;
  int p = t & 63, win = t >> 6;
  int nw = res >> 3;
  int wy = win / nw, wx = win % nw;
  int sy = (wy * 8 + (p >> 3) + shift) & (res - 1);
  int sx = (wx * 8 + (p & 7) + shift) & (res - 1);
  out[i] = (bf16)x[((size_t)sy * res + sx) * C + c];
}

__global__ void k_winmerge_add(float* __restrict__ X, const float* __restrict__ proj,
                               int res, int C, int shift, int total) {
  int i = blockIdx.x * 256 + threadIdx.x;
  if (i >= total) return;
  int c = i % C, pix = i / C;
  int y = pix / res, x = pix % res;
  int yr = (y - shift + res) & (res - 1);
  int xr = (x - shift + res) & (res - 1);
  int nw = res >> 3;
  int win = (yr >> 3) * nw + (xr >> 3);
  int p = ((yr & 7) << 3) + (xr & 7);
  X[i] += proj[((size_t)win * 64 + p) * C + c];
}

// depthwise 3x3 + bias + GELU, bf16 out (feeds leff2 GEMM)
__global__ void k_dwconv(const float* __restrict__ x, const float* __restrict__ w, const float* __restrict__ b,
                         bf16* __restrict__ out, int res, int C, int total) {
  int i = blockIdx.x * 256 + threadIdx.x;
  if (i >= total) return;
  int c = i % C, pix = i / C;
  int y = pix / res, xx = pix % res;
  float acc = b[c];
  for (int ky = 0; ky < 3; ++ky) {
    int iy = y + ky - 1; if (iy < 0 || iy >= res) continue;
    for (int kx = 0; kx < 3; ++kx) {
      int ix = xx + kx - 1; if (ix < 0 || ix >= res) continue;
      acc += x[((size_t)iy * res + ix) * C + c] * w[(ky * 3 + kx) * C + c];
    }
  }
  out[i] = (bf16)gelu_f(acc);
}

__global__ void k_im2col_down(const float* __restrict__ x, bf16* __restrict__ out, int res, int C, int total) {
  int i = blockIdx.x * 256 + threadIdx.x;
  if (i >= total) return;
  int c = i % C, t = i / C;
  int k = t & 15, row = t >> 4;
  int kx = k & 3, ky = k >> 2;
  int half = res >> 1;
  int oy = row / half, ox = row % half;
  int iy = oy * 2 - 1 + ky, ix = ox * 2 - 1 + kx;
  float v = (iy >= 0 && iy < res && ix >= 0 && ix < res) ? x[((size_t)iy * res + ix) * C + c] : 0.f;
  out[i] = (bf16)v;
}

__global__ void k_im2col3(const float* __restrict__ x, bf16* __restrict__ out, int res, int C, int total) {
  int i = blockIdx.x * 256 + threadIdx.x;
  if (i >= total) return;
  int c = i % C, t = i / C;
  int k = t % 9, row = t / 9;
  int kx = k % 3, ky = k / 3;
  int y = row / res, xx = row % res;
  int iy = y - 1 + ky, ix = xx - 1 + kx;
  float v = (iy >= 0 && iy < res && ix >= 0 && ix < res) ? x[((size_t)iy * res + ix) * C + c] : 0.f;
  out[i] = (bf16)v;
}

// up_w (prev, dd2, 2, 2) -> 4 transposed mats[ij] (dd2 x prev) bf16
__global__ void k_pack_upw(const float* __restrict__ w, bf16* __restrict__ out, int prev, int dd2) {
  int total = 4 * prev * dd2;
  int i = blockIdx.x * 256 + threadIdx.x;
  if (i >= total) return;
  int ij = i / (prev * dd2);
  int rem = i % (prev * dd2);
  int d = rem / prev, c = rem % prev;
  out[i] = (bf16)w[((size_t)c * dd2 + d) * 4 + ij];
}

// out_w (3,3,64,2) -> transposed padded (32 x 576) bf16
__global__ void k_pack_outw(const float* __restrict__ w, bf16* __restrict__ out) {
  int i = blockIdx.x * 256 + threadIdx.x;   // 32*576
  if (i >= 32 * 576) return;
  int n = i / 576, r = i % 576;
  out[i] = (n < 2) ? (bf16)w[r * 2 + n] : (bf16)0.f;
}

__global__ void k_upscatter(const float* __restrict__ tmp, float* __restrict__ out,
                            int res, int dd2, int dd, int ii, int jj, int total) {
  int i = blockIdx.x * 256 + threadIdx.x;
  if (i >= total) return;
  int d = i % dd2, pix = i / dd2;
  int y = pix / res, x = pix % res;
  int res2 = res * 2;
  out[((size_t)(2 * y + ii) * res2 + 2 * x + jj) * dd + d] = tmp[i];
}

__global__ void k_copy_cols(const float* __restrict__ src, float* __restrict__ out,
                            int dd2, int dd, int total) {
  int i = blockIdx.x * 256 + threadIdx.x;
  if (i >= total) return;
  int d = i % dd2, row = i / dd2;
  out[(size_t)row * dd + dd2 + d] = src[i];
}

__global__ void k_final(const float* __restrict__ xn, const float* __restrict__ o32,
                        const float* __restrict__ ob, const float* __restrict__ st,
                        float* __restrict__ out) {
  int i = blockIdx.x * 256 + threadIdx.x;
  if (i >= 65536) return;
  for (int c = 0; c < 2; ++c)
    out[i * 2 + c] = (xn[i * 2 + c] + o32[(size_t)i * 32 + c] + ob[c]) * st[2 + c] + st[c];
}

// ================================================================ host orchestration
struct BlockP {
  const float *dw_b, *dw_w, *leff1_b, *leff1_w, *leff2_b, *leff2_w;
  const float *ln1_b, *ln1_g, *ln2_b, *ln2_g, *proj_b, *proj_w, *qkv_b, *qkv_w, *rpb;
};

static inline int cdiv(long long a, int b) { return (int)((a + b - 1) / b); }

extern "C" void kernel_launch(void* const* d_in, const int* in_sizes, int n_in,
                              void* d_out, int out_size, void* d_ws, size_t ws_size,
                              hipStream_t stream) {
  (void)in_sizes; (void)n_in; (void)out_size; (void)ws_size;
  const float* ksp  = (const float*)d_in[0];
  const float* sens = (const float*)d_in[1];
  float* out = (float*)d_out;

  // -------- param pytree walk (sorted-key order: bottleneck, dec, enc, in_b, in_w, out_b, out_w)
  int pi = 2;
  auto nextp = [&]() { return (const float*)d_in[pi++]; };
  auto readBlock = [&]() {
    BlockP p;
    p.dw_b = nextp(); p.dw_w = nextp();
    p.leff1_b = nextp(); p.leff1_w = nextp();
    p.leff2_b = nextp(); p.leff2_w = nextp();
    p.ln1_b = nextp(); p.ln1_g = nextp();
    p.ln2_b = nextp(); p.ln2_g = nextp();
    p.proj_b = nextp(); p.proj_w = nextp();
    p.qkv_b = nextp(); p.qkv_w = nextp();
    p.rpb = nextp();
    return p;
  };
  BlockP bot[2];
  bot[0] = readBlock(); bot[1] = readBlock();
  struct DecP { BlockP b[2]; const float *up_b, *up_w; } dec[4];
  for (int i = 0; i < 4; ++i) { dec[i].b[0] = readBlock(); dec[i].b[1] = readBlock(); dec[i].up_b = nextp(); dec[i].up_w = nextp(); }
  struct EncP { BlockP b[2]; const float *down_b, *down_w; } enc[4];
  for (int i = 0; i < 4; ++i) { enc[i].b[0] = readBlock(); enc[i].b[1] = readBlock(); enc[i].down_b = nextp(); enc[i].down_w = nextp(); }
  const float* in_b = nextp();  const float* in_w = nextp();
  const float* out_b = nextp(); const float* out_w = nextp();

  // -------- workspace carve
  char* base = (char*)d_ws;
  size_t off = 0;
  auto carve = [&](size_t bytes) { void* p = base + off; off = (off + bytes + 255) & ~(size_t)255; return p; };
  float* fftA  = (float*)carve((size_t)12 * 65536 * 2 * 4);
  float* fftB  = (float*)carve((size_t)12 * 65536 * 2 * 4);
  float* img   = (float*)carve((size_t)65536 * 2 * 4);
  float* stats = (float*)carve(256);
  float* xn    = (float*)carve((size_t)65536 * 2 * 4);
  float* X     = (float*)carve((size_t)65536 * 64 * 4);
  float* RA    = (float*)carve((size_t)80 << 20);
  float* RB    = (float*)carve((size_t)64 << 20);
  bf16*  RC    = (bf16*) carve((size_t)34 << 20);
  float* skips[4];
  skips[0] = (float*)carve((size_t)65536 * 32 * 4);
  skips[1] = (float*)carve((size_t)16384 * 64 * 4);
  skips[2] = (float*)carve((size_t)4096 * 128 * 4);
  skips[3] = (float*)carve((size_t)1024 * 256 * 4);
  bf16* arena = (bf16*)carve((size_t)64 << 20);

  size_t acur = 0;
  auto aalloc = [&](size_t n) { bf16* p = arena + acur; acur += (n + 127) & ~(size_t)127; return p; };
  // convert weight (K x N) -> transposed bf16 (N x K)
  auto conv_wt = [&](const float* src, int K, int N) {
    bf16* d = aalloc((size_t)K * N);
    k_f2bt<<<cdiv((long long)K * N, 256), 256, 0, stream>>>(src, d, K, N, K * N);
    return d;
  };

  const int ENC_HEADS[4] = {1, 2, 4, 8}, DEC_HEADS[4] = {16, 8, 4, 2};

  // -------- front end: iFFT2c + coil reduce + norm + input conv
  k_ifft_x<<<12 * 256, 128, 0, stream>>>(ksp, fftA);
  k_ifft_y<<<12 * 256, 128, 0, stream>>>(fftA, fftB);
  k_coil_reduce<<<cdiv(65536, 256), 256, 0, stream>>>(fftB, sens, img);
  k_stats<<<1, 1024, 0, stream>>>(img, stats);
  k_xn<<<cdiv(131072, 256), 256, 0, stream>>>(img, stats, xn);
  k_conv_in<<<cdiv(65536 * 32, 256), 256, 0, stream>>>(xn, in_w, in_b, X);

  // -------- LeWin transformer block
  auto run_block = [&](const BlockP& P, int res, int C, int heads, int shift) {
    int L = res * res, C3 = 3 * C, Ch = 4 * C;
    // ln1 -> RA (f32), window partition (with roll) -> RC (bf16)
    k_layernorm<<<L / 8, 256, 0, stream>>>(X, P.ln1_g, P.ln1_b, RA, L, C);
    k_winpart<<<cdiv((long long)L * C, 256), 256, 0, stream>>>(RA, RC, res, C, shift, L * C);
    // qkv GEMM -> RA (L, 3C)
    bf16* qkvw = conv_wt(P.qkv_w, C, C3);
    { dim3 g(L / 128, C3 / 32); k_gemm<<<g, 128, 0, stream>>>(RC, qkvw, P.qkv_b, nullptr, RA, L, C3, C, 0); }
    // attention -> RC (bf16, feeds proj GEMM directly)
    { dim3 g(L / 64, heads); k_attn<<<g, 128, 0, stream>>>(RA, P.rpb, RC, C, heads, res, shift); }
    // proj GEMM -> RA, merge (+shortcut) into X
    bf16* projw = conv_wt(P.proj_w, C, C);
    { dim3 g(L / 128, C / 32); k_gemm<<<g, 128, 0, stream>>>(RC, projw, P.proj_b, nullptr, RA, L, C, C, 0); }
    k_winmerge_add<<<cdiv((long long)L * C, 256), 256, 0, stream>>>(X, RA, res, C, shift, L * C);
    // LeFF: ln2 (bf16) -> GEMM(C->4C)+GELU -> dwconv+GELU (bf16) -> GEMM(4C->C)+bias+residual
    k_layernorm_b<<<L / 8, 256, 0, stream>>>(X, P.ln2_g, P.ln2_b, RC, L, C);
    bf16* l1w = conv_wt(P.leff1_w, C, Ch);
    { dim3 g(L / 128, Ch / 32); k_gemm<<<g, 128, 0, stream>>>(RC, l1w, P.leff1_b, nullptr, RA, L, Ch, C, 1); }
    k_dwconv<<<cdiv((long long)L * Ch, 256), 256, 0, stream>>>(RA, P.dw_w, P.dw_b, RC, res, Ch, L * Ch);
    bf16* l2w = conv_wt(P.leff2_w, Ch, C);
    { dim3 g(L / 128, C / 32); k_gemm<<<g, 128, 0, stream>>>(RC, l2w, P.leff2_b, X, X, L, C, Ch, 0); }
  };
  auto run_stage = [&](const BlockP* blocks, int res, int C, int heads) {
    run_block(blocks[0], res, C, heads, 0);
    run_block(blocks[1], res, C, heads, 4);
  };

  // -------- encoder
  int res = 256, C = 32;
  for (int si = 0; si < 4; ++si) {
    run_stage(enc[si].b, res, C, ENC_HEADS[si]);
    int L = res * res;
    (void)hipMemcpyAsync(skips[si], X, (size_t)L * C * 4, hipMemcpyDeviceToDevice, stream);
    int Md = L / 4, Kd = 16 * C, Nd = 2 * C;
    k_im2col_down<<<cdiv((long long)Md * Kd, 256), 256, 0, stream>>>(X, RC, res, C, Md * Kd);
    bf16* dw = conv_wt(enc[si].down_w, Kd, Nd);
    { dim3 g(Md / 128, Nd / 32); k_gemm<<<g, 128, 0, stream>>>(RC, dw, enc[si].down_b, nullptr, X, Md, Nd, Kd, 0); }
    res >>= 1; C <<= 1;
  }

  // -------- bottleneck (16x16, C=512)
  run_stage(bot, 16, 512, 16);

  // -------- decoder
  int prev = 512;
  res = 16;
  for (int di = 0; di < 4; ++di) {
    int dd = 512 >> di, dd2 = dd >> 1;
    int Ls = res * res;
    k_f2b<<<cdiv((long long)Ls * prev, 256), 256, 0, stream>>>(X, RC, Ls * prev);
    bf16* upw = aalloc((size_t)4 * prev * dd2);
    k_pack_upw<<<cdiv((long long)4 * prev * dd2, 256), 256, 0, stream>>>(dec[di].up_w, upw, prev, dd2);
    int res2 = res * 2, L2 = res2 * res2;
    for (int ij = 0; ij < 4; ++ij) {
      int ii = ij >> 1, jj = ij & 1;
      dim3 g(Ls / 128, dd2 / 32);
      k_gemm<<<g, 128, 0, stream>>>(RC, upw + (size_t)ij * prev * dd2, dec[di].up_b, nullptr, RA, Ls, dd2, prev, 0);
      k_upscatter<<<cdiv((long long)Ls * dd2, 256), 256, 0, stream>>>(RA, RB, res, dd2, dd, ii, jj, Ls * dd2);
    }
    k_copy_cols<<<cdiv((long long)L2 * dd2, 256), 256, 0, stream>>>(skips[3 - di], RB, dd2, dd, L2 * dd2);
    (void)hipMemcpyAsync(X, RB, (size_t)L2 * dd * 4, hipMemcpyDeviceToDevice, stream);
    res = res2;
    run_stage(dec[di].b, res, dd, DEC_HEADS[di]);
    prev = dd;
  }

  // -------- output conv (3x3, 64->2 via padded N=32 GEMM) + denorm
  {
    int L = 65536, K = 576;
    bf16* colbuf = (bf16*)RA;                 // 75.5 MB bf16 fits RA
    k_im2col3<<<cdiv((long long)L * K, 256), 256, 0, stream>>>(X, colbuf, 256, 64, L * K);
    bf16* ow = aalloc((size_t)32 * K);
    k_pack_outw<<<cdiv(32 * 576, 256), 256, 0, stream>>>(out_w, ow);
    dim3 g(L / 128, 1);
    k_gemm<<<g, 128, 0, stream>>>(colbuf, ow, nullptr, nullptr, RB, L, 32, K, 0);
    k_final<<<cdiv(65536, 256), 256, 0, stream>>>(xn, RB, out_b, stats, out);
  }
}
